// OscillatoryAttention_43885975831132
// MI455X (gfx1250) — matrix operand order repaired and checked by reading, not compiled, not run on hardware
//
#include <hip/hip_runtime.h>
#include <hip/hip_bf16.h>
#include <math.h>

// Flash-attention for: out = softmax((Q+sin_w)(K+cos_w)^T) V
// b=4, t=2048, h=384, fp32 I/O, f16 WMMA with f32 accumulation (CDNA5 wave32).

typedef __attribute__((ext_vector_type(16))) _Float16 v16h;
typedef __attribute__((ext_vector_type(8)))  _Float16 v8h;
typedef __attribute__((ext_vector_type(4)))  _Float16 v4h;
typedef __attribute__((ext_vector_type(2)))  _Float16 v2h;
typedef __attribute__((ext_vector_type(8)))  float    v8f;
typedef __attribute__((ext_vector_type(4)))  float    v4f;

#define T_SEQ    2048
#define HDIM     384
#define MQ       32      // query rows per block
#define KB       64      // key rows per iteration
#define NTHREADS 256

// Padded LDS strides (all 16-byte multiples; chosen so per-lane row steps are
// not 0 mod 64 dword-banks).
#define KSTR   392       // K/Q row stride in f16 (784 B = 196 dwords, 196%64=4)
#define VSTR   72        // Vt row stride in f16 (144 B = 36 dwords)
#define PSTR   72        // P row stride in f16
#define SSTR   68        // S row stride in f32 (272 B = 68 dwords, 68%64=4)

__device__ __forceinline__ v16h cat16(v8h lo, v8h hi) {
  return __builtin_shufflevector(lo, hi, 0,1,2,3,4,5,6,7,8,9,10,11,12,13,14,15);
}

// A-matrix 16x32 f16 fragment from row-major LDS (ISA 7.12.2):
// lane m=lane&15 holds row (row0+m); lanes<16 get K = k0+[0..7],[16..23];
// lanes>=16 get K = k0+[8..15],[24..31].
__device__ __forceinline__ v16h load_fragA(const _Float16* base, int stride,
                                           int row0, int k0, int lane) {
  const int m   = lane & 15;
  const int off = (lane >> 4) ? 8 : 0;
  const _Float16* p = base + (row0 + m) * stride + k0;
  v8h lo = *(const v8h*)(p + off);
  v8h hi = *(const v8h*)(p + 16 + off);
  return cat16(lo, hi);
}

// B-matrix 32x16 f16 fragment. "base" is laid out so that B column n is a
// contiguous row: element [n][k] at base[(row0+n)*stride + k].
// lane n=lane&15 holds column (row0+n); lanes<16 K=k0+[0..7],[16..23], lanes>=16 K=k0+[8..15],[24..31].
__device__ __forceinline__ v16h load_fragB(const _Float16* base, int stride,
                                           int row0, int k0, int lane) {
  const int n   = lane & 15;
  const int off = (lane >> 4) ? 8 : 0;
  const _Float16* p = base + (row0 + n) * stride + k0 + off;
  v8h lo = *(const v8h*)(p);
  v8h hi = *(const v8h*)(p + 16);
  return cat16(lo, hi);
}

__global__ __launch_bounds__(NTHREADS, 1)
void OscillatoryAttention_43885975831132_kernel(const float* __restrict__ Q,
                                                const float* __restrict__ K,
                                                const float* __restrict__ V,
                                                float* __restrict__ O)
{
  __shared__ _Float16 Klds[KB * KSTR];      // keys row-major (+cos), f16
  __shared__ _Float16 Vtlds[HDIM * VSTR];   // V transposed [feat][key], f16
  __shared__ _Float16 Qlds[MQ * KSTR];      // queries (+sin), f16
  __shared__ float    Slds[MQ * SSTR];      // raw scores f32
  __shared__ _Float16 Plds[MQ * PSTR];      // exp'd probs f16
  __shared__ float    coswTab[HDIM];        // cos(c/384) table
  __shared__ float    pm[MQ * 8];           // partial row max
  __shared__ float    ps[MQ * 8];           // partial row sum
  __shared__ float    mArr[MQ], lArr[MQ], mnewArr[MQ], scaleArr[MQ];

  const int tid  = threadIdx.x;
  const int lane = tid & 31;
  const int w    = tid >> 5;                // wave id 0..7
  const int qt   = blockIdx.x % (T_SEQ / MQ);
  const int b    = blockIdx.x / (T_SEQ / MQ);
  const float invH = 1.0f / 384.0f;         // pos_enc/d_model = (2c)/768

  const float* Qg = Q + (size_t)(b * T_SEQ + qt * MQ) * HDIM;
  const float* Kg = K + (size_t)b * T_SEQ * HDIM;
  const float* Vg = V + (size_t)b * T_SEQ * HDIM;
  float*       Og = O + (size_t)(b * T_SEQ + qt * MQ) * HDIM;

  if (tid < MQ) { mArr[tid] = -3.0e38f; lArr[tid] = 0.0f; }
  for (int c = tid; c < HDIM; c += NTHREADS) coswTab[c] = __cosf((float)c * invH);

  // Load query tile (+sin), convert to f16. wave = row group, lanes*4 = cols.
  #pragma unroll
  for (int r0 = 0; r0 < MQ; r0 += 8) {
    const int r = r0 + w;
    const float* src = Qg + (size_t)r * HDIM;
    #pragma unroll
    for (int s = 0; s < 3; ++s) {
      const int cc = s * 128 + lane * 4;
      v4f q = *(const v4f*)(src + cc);
      v4h h;
      #pragma unroll
      for (int j = 0; j < 4; ++j)
        h[j] = (_Float16)(q[j] + __sinf((float)(cc + j) * invH));
      *(v4h*)&Qlds[r * KSTR + cc] = h;
    }
  }
  __syncthreads();

  v8f acc[2][3];
  const v8f vzero = {0.f,0.f,0.f,0.f,0.f,0.f,0.f,0.f};
  #pragma unroll
  for (int qr = 0; qr < 2; ++qr)
    #pragma unroll
    for (int c = 0; c < 3; ++c) acc[qr][c] = vzero;

  const int rsoft = tid >> 3;   // softmax row 0..31
  const int seg   = tid & 7;    // 8-col segment 0..7

  for (int kb = 0; kb < T_SEQ / KB; ++kb) {
    __syncthreads();            // previous PV finished reading Klds/Vtlds
    const int kbase = kb * KB;

    // --- Stage K (+cos) row-major and V transposed into LDS as f16 ---
    // Wave w owns key-row pair (2w, 2w+1) of each 16-row group, so two keys
    // at the same feature pack into one b32 store for the V transpose.
    #pragma unroll
    for (int r0 = 0; r0 < KB; r0 += 16) {
      const int ra = r0 + 2 * w;
      const float* ksrcA = Kg + (size_t)(kbase + ra) * HDIM;
      const float* vsrcA = Vg + (size_t)(kbase + ra) * HDIM;
      #pragma unroll
      for (int s = 0; s < 3; ++s) {
        const int cc = s * 128 + lane * 4;
        v4f ka = *(const v4f*)(ksrcA + cc);
        v4f kq2 = *(const v4f*)(ksrcA + HDIM + cc);
        v4f va = *(const v4f*)(vsrcA + cc);
        v4f vb = *(const v4f*)(vsrcA + HDIM + cc);
        v4f cw = *(const v4f*)&coswTab[cc];
        v4h ha, hb;
        #pragma unroll
        for (int j = 0; j < 4; ++j) {
          ha[j] = (_Float16)(ka[j] + cw[j]);
          hb[j] = (_Float16)(kq2[j] + cw[j]);
        }
        *(v4h*)&Klds[ra * KSTR + cc]       = ha;
        *(v4h*)&Klds[(ra + 1) * KSTR + cc] = hb;
        #pragma unroll
        for (int j = 0; j < 4; ++j) {
          v2h pv2;
          pv2[0] = (_Float16)va[j];
          pv2[1] = (_Float16)vb[j];
          *(v2h*)&Vtlds[(cc + j) * VSTR + ra] = pv2;
        }
      }
    }
    // Prefetch next K/V tile into cache while WMMAs run.
    if (kb + 1 < T_SEQ / KB) {
      const size_t nrow = (size_t)(kbase + KB + (tid >> 2)) * HDIM + (tid & 3) * 96;
      #pragma unroll
      for (int j = 0; j < 3; ++j) {
        __builtin_prefetch(Kg + nrow + j * 32, 0, 1);
        __builtin_prefetch(Vg + nrow + j * 32, 0, 1);
      }
    }
    __syncthreads();

    // --- Scores: wave w computes 16x16 tile (qr = w>>2, kc = w&3) ---
    {
      const int qr = w >> 2, kc = w & 3;
      v8f s = vzero;
      #pragma unroll
      for (int f = 0; f < HDIM / 32; ++f) {
        v16h a  = load_fragA(Qlds, KSTR, qr * 16, f * 32, lane);
        v16h bm = load_fragB(Klds, KSTR, kc * 16, f * 32, lane);
        s = __builtin_amdgcn_wmma_f32_16x16x32_f16(false, a, false, bm,
                                                   (short)0, s, false, false);
      }
      const int rbase = qr * 16 + ((lane >> 4) ? 8 : 0);
      const int col   = kc * 16 + (lane & 15);
      #pragma unroll
      for (int v = 0; v < 8; ++v)
        Slds[(rbase + v) * SSTR + col] = s[v];
    }
    __syncthreads();

    // --- Online softmax: partial max over this thread's 8 columns ---
    v4f s0 = *(const v4f*)&Slds[rsoft * SSTR + seg * 8];
    v4f s1 = *(const v4f*)&Slds[rsoft * SSTR + seg * 8 + 4];
    {
      float mx = fmaxf(fmaxf(fmaxf(s0[0], s0[1]), fmaxf(s0[2], s0[3])),
                       fmaxf(fmaxf(s1[0], s1[1]), fmaxf(s1[2], s1[3])));
      pm[rsoft * 8 + seg] = mx;
    }
    __syncthreads();
    if (tid < MQ) {
      float rm = pm[tid * 8];
      #pragma unroll
      for (int j = 1; j < 8; ++j) rm = fmaxf(rm, pm[tid * 8 + j]);
      const float mo = mArr[tid];
      const float mn = fmaxf(mo, rm);
      mArr[tid]     = mn;
      mnewArr[tid]  = mn;
      scaleArr[tid] = __expf(mo - mn);
    }
    __syncthreads();
    {
      const float mn = mnewArr[rsoft];
      v8h pv;
      float sum = 0.f;
      #pragma unroll
      for (int j = 0; j < 4; ++j) {
        float e = __expf(s0[j] - mn); sum += e; pv[j] = (_Float16)e;
      }
      #pragma unroll
      for (int j = 0; j < 4; ++j) {
        float e = __expf(s1[j] - mn); sum += e; pv[4 + j] = (_Float16)e;
      }
      *(v8h*)&Plds[rsoft * PSTR + seg * 8] = pv;
      ps[rsoft * 8 + seg] = sum;
    }
    __syncthreads();
    if (tid < MQ) {
      float rs = 0.f;
      #pragma unroll
      for (int j = 0; j < 8; ++j) rs += ps[tid * 8 + j];
      lArr[tid] = lArr[tid] * scaleArr[tid] + rs;
    }

    // --- Rescale accumulators, then PV WMMAs ---
    #pragma unroll
    for (int qr = 0; qr < 2; ++qr) {
      const int rbase = qr * 16 + ((lane >> 4) ? 8 : 0);
      float sc[8];
      #pragma unroll
      for (int v = 0; v < 8; ++v) sc[v] = scaleArr[rbase + v];
      #pragma unroll
      for (int c = 0; c < 3; ++c)
        #pragma unroll
        for (int v = 0; v < 8; ++v) acc[qr][c][v] *= sc[v];
    }
    // P fragments loaded once per kk and shared across all 3 column tiles;
    // each Vt fragment shared across both query-row tiles.
    #pragma unroll
    for (int kk = 0; kk < KB / 32; ++kk) {
      v16h a0 = load_fragA(Plds, PSTR, 0,  kk * 32, lane);
      v16h a1 = load_fragA(Plds, PSTR, 16, kk * 32, lane);
      #pragma unroll
      for (int c = 0; c < 3; ++c) {
        const int ct = w * 3 + c;   // output column tile (24 total / 8 waves)
        v16h bm = load_fragB(Vtlds, VSTR, ct * 16, kk * 32, lane);
        acc[0][c] = __builtin_amdgcn_wmma_f32_16x16x32_f16(
            false, a0, false, bm, (short)0, acc[0][c], false, false);
        acc[1][c] = __builtin_amdgcn_wmma_f32_16x16x32_f16(
            false, a1, false, bm, (short)0, acc[1][c], false, false);
      }
    }
  }

  __syncthreads();  // lArr finalized
  #pragma unroll
  for (int qr = 0; qr < 2; ++qr) {
    const int rbase = qr * 16 + ((lane >> 4) ? 8 : 0);
    float linv[8];
    #pragma unroll
    for (int v = 0; v < 8; ++v) linv[v] = 1.0f / lArr[rbase + v];
    #pragma unroll
    for (int c = 0; c < 3; ++c) {
      const int ct  = w * 3 + c;
      const int col = ct * 16 + (lane & 15);
      #pragma unroll
      for (int v = 0; v < 8; ++v)
        Og[(size_t)(rbase + v) * HDIM + col] = acc[qr][c][v] * linv[v];
    }
  }
}

extern "C" void kernel_launch(void* const* d_in, const int* in_sizes, int n_in,
                              void* d_out, int out_size, void* d_ws, size_t ws_size,
                              hipStream_t stream) {
  (void)in_sizes; (void)n_in; (void)d_ws; (void)ws_size; (void)out_size;
  const float* Q = (const float*)d_in[0];
  const float* K = (const float*)d_in[1];
  const float* V = (const float*)d_in[2];
  float* O = (float*)d_out;
  dim3 grid(4 * (T_SEQ / MQ));   // 256 blocks: (batch, 32-row query tile)
  dim3 block(NTHREADS);
  hipLaunchKernelGGL(OscillatoryAttention_43885975831132_kernel,
                     grid, block, 0, stream, Q, K, V, O);
}
